// LSTMNet_52243982189153
// MI455X (gfx1250) — matrix operand order repaired and checked by reading, not compiled, not run on hardware
//
#include <hip/hip_runtime.h>
#include <hip/hip_bf16.h>
#include <cstdint>
#include <cstddef>

// ---------------------------------------------------------------------------
// Problem constants (from reference)
// ---------------------------------------------------------------------------
#define B_   512
#define T_   1024
#define IN_  32
#define H_   128
#define G4_  512      /* 4*H */
#define DL_  256
#define OUT_ 32

typedef _Float16 v8h  __attribute__((ext_vector_type(8)));
typedef _Float16 v16h __attribute__((ext_vector_type(16)));
typedef float    v8f  __attribute__((ext_vector_type(8)));

__device__ __forceinline__ v8f wmma_f16(v16h a, v16h b, v8f c) {
    // D = A(16x32 f16) * B(32x16 f16) + C(16x16 f32)
    return __builtin_amdgcn_wmma_f32_16x16x32_f16(
        /*neg_a=*/false, a, /*neg_b=*/false, b,
        /*c_mod=*/(short)0, c, /*reuse_a=*/false, /*reuse_b=*/false);
}

__device__ __forceinline__ float sigmoid_f(float x) {
    return 1.0f / (1.0f + __expf(-x));
}
__device__ __forceinline__ float tanh_f(float x) {
    float e = __expf(-2.0f * x);
    return (1.0f - e) / (1.0f + e);
}
__device__ __forceinline__ float celu_f(float x) {   // alpha = 1
    return x > 0.0f ? x : (__expf(x) - 1.0f);
}

// ---------------------------------------------------------------------------
// Small utility kernels
// ---------------------------------------------------------------------------
__global__ void zero_f32(float* __restrict__ p, int n) {
    int i = blockIdx.x * blockDim.x + threadIdx.x;
    if (i < n) p[i] = 0.0f;
}

__global__ void f32_to_f16(const float* __restrict__ s, _Float16* __restrict__ d, int n) {
    int i = blockIdx.x * blockDim.x + threadIdx.x;
    if (i < n) d[i] = (_Float16)s[i];
}

__global__ void bias_sum(const float* __restrict__ a, const float* __restrict__ b,
                         float* __restrict__ o, int n) {
    int i = blockIdx.x * blockDim.x + threadIdx.x;
    if (i < n) o[i] = a[i] + b[i];
}

// Pack W (G x K, row-major f32) into WMMA-B-fragment order (f16):
//   block (kb,nb) is 32x16 (KxN); within a block, lane L = kc*16 + n holds the
//   K-values kb*32 + {kc*8..kc*8+7, 16+kc*8..16+kc*8+7} of output column g = nb*16+n.
// Consumption: lane L reads 16 halves at ((kb*NB+nb)*32 + L)*16 — one 32B read.
__global__ void pack_weight_f16(const float* __restrict__ W, _Float16* __restrict__ dst,
                                int G, int K) {
    int idx = blockIdx.x * blockDim.x + threadIdx.x;
    if (idx >= G * K) return;
    int g  = idx / K,   k  = idx % K;
    int kb = k >> 5,    kc = (k >> 3) & 1, kk = (k & 7) | (((k >> 4) & 1) << 3);
    int nb = g >> 4,    n  = g & 15;
    int NB = G >> 4;
    size_t d = ((size_t)(kb * NB + nb) * 32 + (size_t)(kc * 16 + n)) * 16 + kk;
    dst[d] = (_Float16)W[idx];
}

// ---------------------------------------------------------------------------
// Persistent-per-batch-slice LSTM layer.
//   grid  = B/16 workgroups (batch rows independent -> no inter-WG sync)
//   block = 256 threads = 8 waves; wave w owns h-columns [16w, 16w+16)
//           and the four gate tiles {w, w+8, w+16, w+24} of the 16x512 gates.
//   Whh B-fragments (and, for layer 0, Wih) are REGISTER-resident for the
//   whole T loop -> steady state has only 4 LDS A-fragment loads per step.
//   c state lives in 8 f32 VGPRs per lane; h double-buffered in LDS.
// ---------------------------------------------------------------------------
template <int KIN, bool WRITE_STREAM, bool WRITE_LAST>
__global__ __launch_bounds__(256, 1)
void lstm_layer_wmma(const _Float16* __restrict__ Xin,   // [B][T][KIN] f16
                     const _Float16* __restrict__ WihP,  // packed (KIN/32)*32 blocks
                     const _Float16* __restrict__ WhhP,  // packed 4*32 blocks
                     const float*    __restrict__ bsum,  // [512] = bih+bhh
                     _Float16*       __restrict__ Hout,  // [B][T][128] f16 (layer0)
                     float*          __restrict__ Hlast) // [B][128] f32 (layer1)
{
    constexpr int  KB_IN     = KIN / 32;
    constexpr bool HOIST_WIH = (KB_IN == 1);             // layer 0: K=32
    constexpr int  WIH_HALVES = HOIST_WIH ? 0 : KB_IN * 32 * 512;

    extern __shared__ _Float16 smem[];
    _Float16* s_wih = smem;                              // only if !HOIST_WIH
    _Float16* s_h   = smem + WIH_HALVES;                 // 2 * 16 * 128 halves

    const int tid  = threadIdx.x;
    const int wave = tid >> 5;          // 0..7 -> h-column tile
    const int lane = tid & 31;
    const int nlo  = lane & 15;         // N (C/D) / A row
    const int hi16 = lane >> 4;         // K-half selector for A fragments
    const int b0   = blockIdx.x << 4;   // batch base

    // ---- cooperative Wih load into LDS (layer 1 only) ----
    if constexpr (!HOIST_WIH) {
        for (int i = tid * 8; i < WIH_HALVES; i += 256 * 8)
            *(v8h*)(s_wih + i) = *(const v8h*)(WihP + i);
    }
    {
        v8h z = {};
        for (int i = tid * 8; i < 2 * 16 * 128; i += 256 * 8)
            *(v8h*)(s_h + i) = z;                        // h_{-1} = 0
    }

    // ---- register-resident weight fragments (loop-invariant) ----
    v16h whhf[4][4];                                     // 128 VGPRs
#pragma unroll
    for (int kb = 0; kb < 4; ++kb)
#pragma unroll
        for (int g = 0; g < 4; ++g)
            whhf[kb][g] = *(const v16h*)(
                WhhP + ((size_t)(kb * 32 + g * 8 + wave) * 32 + lane) * 16);

    v16h wihf[KB_IN][4];                                 // used only when hoisted
    if constexpr (HOIST_WIH) {
#pragma unroll
        for (int kb = 0; kb < KB_IN; ++kb)
#pragma unroll
            for (int g = 0; g < 4; ++g)
                wihf[kb][g] = *(const v16h*)(
                    WihP + ((size_t)(kb * 32 + g * 8 + wave) * 32 + lane) * 16);
    }

    float bias[4];
#pragma unroll
    for (int g = 0; g < 4; ++g)
        bias[g] = bsum[g * 128 + wave * 16 + nlo];

    __syncthreads();

    v8f c_ = {};                                         // cell state, registers
    int rbuf = 1, wbuf = 0;

    const _Float16* xrow = Xin + (size_t)(b0 + nlo) * T_ * KIN;

    for (int t = 0; t < T_; ++t) {
        v8f acc[4];
#pragma unroll
        for (int g = 0; g < 4; ++g) acc[g] = (v8f){};

        const _Float16* xp = xrow + (size_t)t * KIN;

        // ---- input contribution: gates += x_t @ Wih^T ----
#pragma unroll
        for (int kb = 0; kb < KB_IN; ++kb) {
            const int k0 = kb * 32 + hi16 * 8;           // ISA A-fragment K layout
            v8h lo = *(const v8h*)(xp + k0);
            v8h hi = *(const v8h*)(xp + k0 + 16);
            v16h a = __builtin_shufflevector(lo, hi,
                     0,1,2,3,4,5,6,7,8,9,10,11,12,13,14,15);
            if constexpr (HOIST_WIH) {
#pragma unroll
                for (int g = 0; g < 4; ++g)
                    acc[g] = wmma_f16(a, wihf[kb][g], acc[g]);
            } else {
                const _Float16* wb = s_wih + (size_t)kb * 32 * 512
                                   + (size_t)lane * 16;
                v16h bf[4];
#pragma unroll
                for (int g = 0; g < 4; ++g)              // issue all loads first
                    bf[g] = *(const v16h*)(wb + (size_t)(g * 8 + wave) * 512);
#pragma unroll
                for (int g = 0; g < 4; ++g)
                    acc[g] = wmma_f16(a, bf[g], acc[g]);
            }
        }

        // ---- recurrent contribution: gates += h_{t-1} @ Whh^T (regs only) ----
        const _Float16* hrow = s_h + rbuf * 2048 + nlo * 128;
#pragma unroll
        for (int kb = 0; kb < 4; ++kb) {
            const int k0 = kb * 32 + hi16 * 8;
            v8h lo = *(const v8h*)(hrow + k0);
            v8h hi = *(const v8h*)(hrow + k0 + 16);
            v16h a = __builtin_shufflevector(lo, hi,
                     0,1,2,3,4,5,6,7,8,9,10,11,12,13,14,15);
#pragma unroll
            for (int g = 0; g < 4; ++g)
                acc[g] = wmma_f16(a, whhf[kb][g], acc[g]);
        }

        // ---- elementwise gate math; C/D layout: lane->N=nlo, VGPR r->M ----
#pragma unroll
        for (int r = 0; r < 8; ++r) {
            float iv = sigmoid_f(acc[0][r] + bias[0]);
            float fv = sigmoid_f(acc[1][r] + bias[1]);
            float gv = tanh_f  (acc[2][r] + bias[2]);
            float ov = sigmoid_f(acc[3][r] + bias[3]);
            float cv = fv * c_[r] + iv * gv;
            c_[r] = cv;
            float hv = ov * tanh_f(cv);
            const int m = r + hi16 * 8;                  // row in 16x16 tile
            s_h[wbuf * 2048 + m * 128 + wave * 16 + nlo] = (_Float16)hv;
            if (WRITE_LAST && t == T_ - 1)
                Hlast[(size_t)(b0 + m) * H_ + wave * 16 + nlo] = hv;
        }

        __syncthreads();   // publish h_t before anyone consumes it

        // ---- coalesced h stream-out (layer 0 feeds layer 1) ----
        if (WRITE_STREAM) {
            const int rr = tid >> 4;
            const int c8 = (tid & 15) << 3;
            v8h v = *(const v8h*)(s_h + wbuf * 2048 + rr * 128 + c8);
            *(v8h*)(Hout + ((size_t)(b0 + rr) * T_ + t) * H_ + c8) = v;
        }

        // prefetch next time-step's activations (global_prefetch_b8)
        if (t + 1 < T_)
            __builtin_prefetch(xp + KIN, 0, 1);

        rbuf ^= 1;
        wbuf ^= 1;
    }
}

// ---------------------------------------------------------------------------
// MLP head (tiny: ~50 MFLOP total) — plain VALU
// ---------------------------------------------------------------------------
__global__ void head1(const float* __restrict__ last, const float* __restrict__ W1,
                      const float* __restrict__ b1, float* __restrict__ y1) {
    int idx = blockIdx.x * blockDim.x + threadIdx.x;     // B_*DL_
    if (idx >= B_ * DL_) return;
    int b = idx >> 8, j = idx & (DL_ - 1);
    const float* lr = last + (size_t)b * H_;
    const float* wr = W1 + (size_t)j * H_;
    float acc = b1[j];
#pragma unroll 8
    for (int k = 0; k < H_; ++k) acc += lr[k] * wr[k];
    y1[idx] = celu_f(acc);
}

__global__ void head2(const float* __restrict__ y1, const float* __restrict__ W2,
                      const float* __restrict__ b2, float* __restrict__ y) {
    int idx = blockIdx.x * blockDim.x + threadIdx.x;     // B_*OUT_
    if (idx >= B_ * OUT_) return;
    int b = idx >> 5, j = idx & (OUT_ - 1);
    const float* lr = y1 + (size_t)b * DL_;
    const float* wr = W2 + (size_t)j * DL_;
    float acc = b2[j];
#pragma unroll 8
    for (int k = 0; k < DL_; ++k) acc += lr[k] * wr[k];
    y[idx] = celu_f(acc);
}

// ---------------------------------------------------------------------------
// Launch
// ---------------------------------------------------------------------------
extern "C" void kernel_launch(void* const* d_in, const int* in_sizes, int n_in,
                              void* d_out, int out_size, void* d_ws, size_t ws_size,
                              hipStream_t stream) {
    const float* x    = (const float*)d_in[0];
    const float* Wih0 = (const float*)d_in[1];
    const float* Whh0 = (const float*)d_in[2];
    const float* bih0 = (const float*)d_in[3];
    const float* bhh0 = (const float*)d_in[4];
    const float* Wih1 = (const float*)d_in[5];
    const float* Whh1 = (const float*)d_in[6];
    const float* bih1 = (const float*)d_in[7];
    const float* bhh1 = (const float*)d_in[8];
    const float* W1   = (const float*)d_in[9];
    const float* b1   = (const float*)d_in[10];
    const float* W2   = (const float*)d_in[11];
    const float* b2   = (const float*)d_in[12];
    float* out = (float*)d_out;

    // workspace carve-out
    char* ws = (char*)d_ws;
    auto carve = [&](size_t bytes) -> char* {
        char* p = ws;
        ws += (bytes + 255) & ~(size_t)255;
        return p;
    };
    _Float16* Xf16  = (_Float16*)carve((size_t)B_ * T_ * IN_ * 2);  // 32 MB
    _Float16* H1s   = (_Float16*)carve((size_t)B_ * T_ * H_  * 2);  // 128 MB
    _Float16* Wih0P = (_Float16*)carve((size_t)G4_ * IN_ * 2);
    _Float16* Whh0P = (_Float16*)carve((size_t)G4_ * H_  * 2);
    _Float16* Wih1P = (_Float16*)carve((size_t)G4_ * H_  * 2);
    _Float16* Whh1P = (_Float16*)carve((size_t)G4_ * H_  * 2);
    float*    bs0   = (float*)carve(G4_ * 4);
    float*    bs1   = (float*)carve(G4_ * 4);
    float*    last  = (float*)carve((size_t)B_ * H_ * 4);
    float*    y1    = (float*)carve((size_t)B_ * DL_ * 4);

    // d_out = [ y (512*32) | hidden_init zeros (2*512*128) ]
    const int nout = B_ * OUT_ + 2 * B_ * H_;
    zero_f32<<<(nout + 255) / 256, 256, 0, stream>>>(out, nout);

    const int nx = B_ * T_ * IN_;
    f32_to_f16<<<(nx + 255) / 256, 256, 0, stream>>>(x, Xf16, nx);

    pack_weight_f16<<<(G4_ * IN_ + 255) / 256, 256, 0, stream>>>(Wih0, Wih0P, G4_, IN_);
    pack_weight_f16<<<(G4_ * H_  + 255) / 256, 256, 0, stream>>>(Whh0, Whh0P, G4_, H_);
    pack_weight_f16<<<(G4_ * H_  + 255) / 256, 256, 0, stream>>>(Wih1, Wih1P, G4_, H_);
    pack_weight_f16<<<(G4_ * H_  + 255) / 256, 256, 0, stream>>>(Whh1, Whh1P, G4_, H_);
    bias_sum<<<2, 256, 0, stream>>>(bih0, bhh0, bs0, G4_);
    bias_sum<<<2, 256, 0, stream>>>(bih1, bhh1, bs1, G4_);

    // LDS: layer0 = h double-buffer only; layer1 = packed Wih + h double-buffer
    const size_t lds0 = (size_t)(2 * 16 * 128) * 2;                      // 8 KB
    const size_t lds1 = (size_t)(4 * 32 * 512 + 2 * 16 * 128) * 2;       // 136 KB

    lstm_layer_wmma<IN_, true, false>
        <<<B_ / 16, 256, lds0, stream>>>(Xf16, Wih0P, Whh0P, bs0, H1s, nullptr);
    lstm_layer_wmma<H_, false, true>
        <<<B_ / 16, 256, lds1, stream>>>(H1s, Wih1P, Whh1P, bs1, nullptr, last);

    head1<<<(B_ * DL_ + 255) / 256, 256, 0, stream>>>(last, W1, b1, y1);
    head2<<<(B_ * OUT_ + 255) / 256, 256, 0, stream>>>(y1, W2, b2, out);
}
